// Attention_org_1554778161848
// MI455X (gfx1250) — matrix operand order repaired and checked by reading, not compiled, not run on hardware
//
#include <hip/hip_runtime.h>
#include <hip/hip_bf16.h>
#include <cmath>

// ---------------------------------------------------------------------------
// MI455X (gfx1250, wave32). All heavy math = v_wmma_f32_16x16x32_bf16 with
// f32 accumulate (bf16 is ~8x the f32-WMMA rate; workload is ~290 GFLOP vs
// <1GB HBM traffic so precision choice decides compute-vs-memory bound).
// Staging is vectorized (b128 global loads, packed b64/b128/b32 LDS stores)
// and register-pipelined so global-load latency overlaps WMMA compute.
// f32<->bf16 conversions use native __bf16 casts (HW cvt, RNE) instead of
// integer emulation.
// ---------------------------------------------------------------------------

typedef __attribute__((ext_vector_type(16))) __bf16 bf16x16;
typedef __attribute__((ext_vector_type(8)))  __bf16 bf16x8;
typedef __attribute__((ext_vector_type(8)))  float  f32x8;

__device__ __forceinline__ unsigned short f2bf(float f) {
  union { __bf16 h; unsigned short u; } x;
  x.h = (__bf16)f;                          // native cvt, round-to-nearest-even
  return x.u;
}
__device__ __forceinline__ float bf2f(unsigned short b) {
  union { unsigned short u; __bf16 h; } x;
  x.u = b;
  return (float)x.h;                        // exact widen
}
__device__ __forceinline__ unsigned pk2(unsigned short lo, unsigned short hi) {
  return (unsigned)lo | ((unsigned)hi << 16);
}
__device__ __forceinline__ uint2 pk4(float4 v) {
  return make_uint2(pk2(f2bf(v.x), f2bf(v.y)), pk2(f2bf(v.z), f2bf(v.w)));
}

// ---------------------------------------------------------------------------
// Batched tiled GEMM:  C[z] = alpha * rowS[m] * colS[n] * (A[z] x B[z])
//   A: M x K row-major (k contiguous), f32 or bf16 (ABF)
//   B: BNMAJ ? N x K row-major (QK^T) : K x N row-major (1x1 conv / P*V)
//   C: M x N, f32 or bf16 (OBF)
// 256 threads (8 waves), 64x64 tile, K-step 32, 2-stage register pipeline.
// M,N multiples of 64; K multiple of 32 (true for every shape here).
// ---------------------------------------------------------------------------
template<bool ABF, bool BBF, bool BNMAJ, bool OBF>
__global__ __launch_bounds__(256)
void wmma_gemm(const void* __restrict__ Ap, const void* __restrict__ Bp,
               void* __restrict__ Cp, int Kdim, int lda, int ldb, int ldc,
               long long strideA, long long strideB, long long strideC,
               float alpha,
               const float* __restrict__ rowScale, int rsStride,
               const float* __restrict__ colScale, int csStride)
{
  // +8 ushort pad: 80B row stride (mult of 16) -> aligned b128 chunks,
  // conflict-free 16-lane fragment reads.
  __shared__ __align__(16) unsigned short lA[64][40];
  __shared__ __align__(16) unsigned short lB[64][40];

  const int tid   = threadIdx.x;
  const int z     = blockIdx.z;
  const int mBase = blockIdx.y * 64;
  const int nBase = blockIdx.x * 64;

  const float*          Afz = (const float*)Ap          + (size_t)z * (ABF ? 0 : strideA);
  const unsigned short* Ahz = (const unsigned short*)Ap + (size_t)z * (ABF ? strideA : 0);
  const float*          Bfz = (const float*)Bp          + (size_t)z * (BBF ? 0 : strideB);
  const unsigned short* Bhz = (const unsigned short*)Bp + (size_t)z * (BBF ? strideB : 0);
  float*                Cf  = (float*)Cp;
  unsigned short*       Ch  = (unsigned short*)Cp;

  // ---- staging coordinates (fixed trip counts, vector transfers) ----
  // bf16 rows (A / NT-B): one b128 per thread: row sr8, 8 cols at sc8.
  const int sr8 = tid >> 2, sc8 = (tid & 3) * 8;
  // f32 rows (A / NT-B): two float4 per thread: rows fr, fr+32, 4 cols at fc.
  const int fr = tid >> 3, fc = (tid & 7) * 4;
  // NN-B (k-major): vertical k-pair (kkp,kkp+1) x 4 cols at nc -> 4 b32 stores.
  const int kkp = (tid >> 4) * 2, nc = (tid & 15) * 4;

  // staging registers (only the template-selected ones survive DCE)
  float4 aR0, aR1, bR0, bR1;
  uint4  aRH, bRH;
  uint2  bP0, bP1;

  auto loadA = [&](int k0) {
    if (ABF) {
      aRH = *(const uint4*)(Ahz + (size_t)(mBase + sr8) * lda + (k0 + sc8));
    } else {
      aR0 = *(const float4*)(Afz + (size_t)(mBase + fr) * lda + (k0 + fc));
      aR1 = *(const float4*)(Afz + (size_t)(mBase + fr + 32) * lda + (k0 + fc));
    }
  };
  auto loadB = [&](int k0) {
    if (BNMAJ) {
      if (BBF) {
        bRH = *(const uint4*)(Bhz + (size_t)(nBase + sr8) * ldb + (k0 + sc8));
      } else {
        bR0 = *(const float4*)(Bfz + (size_t)(nBase + fr) * ldb + (k0 + fc));
        bR1 = *(const float4*)(Bfz + (size_t)(nBase + fr + 32) * ldb + (k0 + fc));
      }
    } else {
      if (BBF) {
        bP0 = *(const uint2*)(Bhz + (size_t)(k0 + kkp) * ldb + (nBase + nc));
        bP1 = *(const uint2*)(Bhz + (size_t)(k0 + kkp + 1) * ldb + (nBase + nc));
      } else {
        bR0 = *(const float4*)(Bfz + (size_t)(k0 + kkp) * ldb + (nBase + nc));
        bR1 = *(const float4*)(Bfz + (size_t)(k0 + kkp + 1) * ldb + (nBase + nc));
      }
    }
  };
  auto storeTiles = [&]() {
    if (ABF) {
      *(uint4*)&lA[sr8][sc8] = aRH;
    } else {
      *(uint2*)&lA[fr][fc]      = pk4(aR0);
      *(uint2*)&lA[fr + 32][fc] = pk4(aR1);
    }
    if (BNMAJ) {
      if (BBF) {
        *(uint4*)&lB[sr8][sc8] = bRH;
      } else {
        *(uint2*)&lB[fr][fc]      = pk4(bR0);
        *(uint2*)&lB[fr + 32][fc] = pk4(bR1);
      }
    } else {
      unsigned short p0[4], p1[4];
      if (BBF) {
        p0[0] = (unsigned short)bP0.x; p0[1] = (unsigned short)(bP0.x >> 16);
        p0[2] = (unsigned short)bP0.y; p0[3] = (unsigned short)(bP0.y >> 16);
        p1[0] = (unsigned short)bP1.x; p1[1] = (unsigned short)(bP1.x >> 16);
        p1[2] = (unsigned short)bP1.y; p1[3] = (unsigned short)(bP1.y >> 16);
      } else {
#pragma unroll
        for (int j = 0; j < 4; ++j) {
          p0[j] = f2bf(((const float*)&bR0)[j]);
          p1[j] = f2bf(((const float*)&bR1)[j]);
        }
      }
#pragma unroll
      for (int j = 0; j < 4; ++j)           // k-pair packed per n: ds_store_b32
        *(unsigned*)&lB[nc + j][kkp] = pk2(p0[j], p1[j]);
    }
  };

  // ---- wave/subtile mapping: 8 waves x 2 subtiles = 16 (4m x 4n) ----
  const int wave = tid >> 5, lane = tid & 31;
  const int h = lane >> 4, ln = lane & 15;
  const int mi0    = (wave & 1) * 2;
  const int ni     = wave >> 1;
  const int amRow0 = mi0 * 16 + ln;
  const int amRow1 = amRow0 + 16;
  const int bnRow  = ni * 16 + ln;

  f32x8 acc0 = {}; f32x8 acc1 = {};

  loadA(0); loadB(0);                       // prime the pipeline
  for (int k0 = 0; k0 < Kdim; k0 += 32) {
    storeTiles();                           // regs -> LDS (b128/b64/b32 stores)
    __syncthreads();
    if (k0 + 32 < Kdim) {                   // next tile's loads fly during WMMA
      loadA(k0 + 32);
      loadB(k0 + 32);
    }

    // fragments per ISA 7.12.2 layouts
    bf16x16 a0, a1, bF;
    ((bf16x8*)&a0)[0] = *(const bf16x8*)&lA[amRow0][h * 8];
    ((bf16x8*)&a0)[1] = *(const bf16x8*)&lA[amRow0][16 + h * 8];
    ((bf16x8*)&a1)[0] = *(const bf16x8*)&lA[amRow1][h * 8];
    ((bf16x8*)&a1)[1] = *(const bf16x8*)&lA[amRow1][16 + h * 8];
    ((bf16x8*)&bF)[0] = *(const bf16x8*)&lB[bnRow][h * 8];
    ((bf16x8*)&bF)[1] = *(const bf16x8*)&lB[bnRow][16 + h * 8];

    acc0 = __builtin_amdgcn_wmma_f32_16x16x32_bf16(false, a0, false, bF, (short)0, acc0, false, false);
    acc1 = __builtin_amdgcn_wmma_f32_16x16x32_bf16(false, a1, false, bF, (short)0, acc1, false, false);

    __syncthreads();
  }

  // ---- epilogue: alpha * rowScale[m] * colScale[n] ----
  const int gn = nBase + ni * 16 + ln;
  const float cs = colScale ? colScale[(size_t)z * csStride + gn] : 1.0f;
#pragma unroll
  for (int r = 0; r < 8; ++r) {
    int m0 = mBase + mi0 * 16 + r + 8 * h;  // D: VGPR r <-> row r + 8*half
    int m1 = m0 + 16;
    float rs0 = rowScale ? rowScale[(size_t)z * rsStride + m0] : 1.0f;
    float rs1 = rowScale ? rowScale[(size_t)z * rsStride + m1] : 1.0f;
    float v0 = acc0[r] * alpha * rs0 * cs;
    float v1 = acc1[r] * alpha * rs1 * cs;
    size_t o0 = (size_t)z * strideC + (size_t)m0 * ldc + gn;
    size_t o1 = (size_t)z * strideC + (size_t)m1 * ldc + gn;
    if (OBF) { Ch[o0] = f2bf(v0); Ch[o1] = f2bf(v1); }
    else     { Cf[o0] = v0;       Cf[o1] = v1; }
  }
}

// ---------------------------------------------------------------------------
// Depthwise (npg=1) / grouped-2 (npg=2) 3x3 SAME conv on one 64x64 channel
// per block, LDS-staged image, plus per-(b,c) reciprocal L2 norm.
// ---------------------------------------------------------------------------
__global__ __launch_bounds__(256)
void dw3x3_norm(const unsigned short* __restrict__ X, const float* __restrict__ W,
                unsigned short* __restrict__ Z, float* __restrict__ rnorm,
                int C, int npg)
{
  __shared__ float tile[2][4096];
  __shared__ float red[256];
  const int tid = threadIdx.x;
  const int b = blockIdx.x / C;
  const int c = blockIdx.x - b * C;
  const int icBase = (npg == 2) ? (c & ~1) : c;

  for (int g = 0; g < npg; ++g) {
    const unsigned short* src = X + ((size_t)b * C + icBase + g) * 4096;
#pragma unroll
    for (int j = 0; j < 2; ++j) {
      int p = (tid + j * 256) * 8;
      uint4 v = *(const uint4*)(src + p);           // b128 load, 8 bf16
      tile[g][p + 0] = bf2f((unsigned short)v.x);
      tile[g][p + 1] = bf2f((unsigned short)(v.x >> 16));
      tile[g][p + 2] = bf2f((unsigned short)v.y);
      tile[g][p + 3] = bf2f((unsigned short)(v.y >> 16));
      tile[g][p + 4] = bf2f((unsigned short)v.z);
      tile[g][p + 5] = bf2f((unsigned short)(v.z >> 16));
      tile[g][p + 6] = bf2f((unsigned short)v.w);
      tile[g][p + 7] = bf2f((unsigned short)(v.w >> 16));
    }
  }
  float wreg[2][9];
  for (int g = 0; g < npg; ++g)
#pragma unroll
    for (int i = 0; i < 9; ++i) wreg[g][i] = W[((size_t)c * npg + g) * 9 + i];
  __syncthreads();

  unsigned short* zrow = Z + ((size_t)b * C + c) * 4096;
  float ss = 0.0f;
  for (int i = 0; i < 16; ++i) {
    int p = tid + i * 256;
    int y = p >> 6, x = p & 63;
    float acc = 0.0f;
    for (int g = 0; g < npg; ++g) {
#pragma unroll
      for (int dy = -1; dy <= 1; ++dy) {
        int yy = y + dy;
        if (yy < 0 || yy > 63) continue;
#pragma unroll
        for (int dx = -1; dx <= 1; ++dx) {
          int xx = x + dx;
          if (xx < 0 || xx > 63) continue;
          acc += tile[g][yy * 64 + xx] * wreg[g][(dy + 1) * 3 + (dx + 1)];
        }
      }
    }
    zrow[p] = f2bf(acc);
    ss += acc * acc;
  }
  red[tid] = ss; __syncthreads();
  for (int s = 128; s > 0; s >>= 1) {
    if (tid < s) red[tid] += red[tid + s];
    __syncthreads();
  }
  if (tid == 0) rnorm[(size_t)b * C + c] = 1.0f / fmaxf(sqrtf(red[0]), 1e-12f);
}

// InstanceNorm stats over the whole c x KV attn matrix, one block per batch.
__global__ __launch_bounds__(256)
void inorm_stats(const float* __restrict__ S, long long n, float* __restrict__ stats)
{
  __shared__ float r1[256], r2[256];
  const float4* base = (const float4*)(S + (size_t)blockIdx.x * n);
  const long long n4 = n >> 2;
  float s = 0.0f, s2 = 0.0f;
  for (long long i = threadIdx.x; i < n4; i += 256) {
    float4 v = base[i];
    s  += v.x + v.y + v.z + v.w;
    s2 += v.x * v.x + v.y * v.y + v.z * v.z + v.w * v.w;
  }
  r1[threadIdx.x] = s; r2[threadIdx.x] = s2; __syncthreads();
  for (int st = 128; st > 0; st >>= 1) {
    if (threadIdx.x < st) { r1[threadIdx.x] += r1[threadIdx.x + st]; r2[threadIdx.x] += r2[threadIdx.x + st]; }
    __syncthreads();
  }
  if (threadIdx.x == 0) {
    float mu  = r1[0] / (float)n;
    float var = fmaxf(r2[0] / (float)n - mu * mu, 0.0f);
    stats[blockIdx.x * 2]     = mu;
    stats[blockIdx.x * 2 + 1] = rsqrtf(var + 1e-5f);
  }
}

// Instance-norm then row softmax over KV=960; bf16 probs out. Block per (b,row).
__global__ __launch_bounds__(256)
void inorm_softmax(const float* __restrict__ S, const float* __restrict__ stats,
                   unsigned short* __restrict__ P, int c)
{
  __shared__ float red[256];
  const int b   = blockIdx.x / c;
  const int row = blockIdx.x - b * c;
  const float* srow = S + ((size_t)b * c + row) * 960;
  unsigned short* prow = P + ((size_t)b * c + row) * 960;
  const float mu = stats[b * 2];
  const float rs = stats[b * 2 + 1];

  float x[4];
  float mx = -3.4e38f;
#pragma unroll
  for (int k = 0; k < 4; ++k) {
    int j = threadIdx.x + k * 256;
    x[k] = (j < 960) ? (srow[j] - mu) * rs : -3.4e38f;
    mx = fmaxf(mx, x[k]);
  }
  red[threadIdx.x] = mx; __syncthreads();
  for (int st = 128; st > 0; st >>= 1) {
    if (threadIdx.x < st) red[threadIdx.x] = fmaxf(red[threadIdx.x], red[threadIdx.x + st]);
    __syncthreads();
  }
  mx = red[0]; __syncthreads();

  float e[4]; float sum = 0.0f;
#pragma unroll
  for (int k = 0; k < 4; ++k) {
    int j = threadIdx.x + k * 256;
    e[k] = (j < 960) ? __expf(x[k] - mx) : 0.0f;
    sum += e[k];
  }
  red[threadIdx.x] = sum; __syncthreads();
  for (int st = 128; st > 0; st >>= 1) {
    if (threadIdx.x < st) red[threadIdx.x] += red[threadIdx.x + st];
    __syncthreads();
  }
  const float inv = 1.0f / red[0];
#pragma unroll
  for (int k = 0; k < 4; ++k) {
    int j = threadIdx.x + k * 256;
    if (j < 960) prow[j] = f2bf(e[k] * inv);
  }
}

// ---------------------------------------------------------------------------
extern "C" void kernel_launch(void* const* d_in, const int* in_sizes, int n_in,
                              void* d_out, int out_size, void* d_ws, size_t ws_size,
                              hipStream_t stream)
{
  (void)in_sizes; (void)n_in; (void)out_size; (void)ws_size;
  const int CNv[4] = {64, 128, 256, 512};
  const int B = 8, HW = 4096, KV = 960;

  const float* emb[4]  = {(const float*)d_in[0], (const float*)d_in[1],
                          (const float*)d_in[2], (const float*)d_in[3]};
  const float* emb_all = (const float*)d_in[4];
  const float* wm[4], *wq[4], *wp[4];
  for (int i = 0; i < 4; ++i) {
    wm[i] = (const float*)d_in[5 + 3 * i];
    wq[i] = (const float*)d_in[6 + 3 * i];
    wp[i] = (const float*)d_in[7 + 3 * i];
  }
  const float* wmk = (const float*)d_in[17];
  const float* wmv = (const float*)d_in[18];
  const float* wk  = (const float*)d_in[19];
  const float* wv  = (const float*)d_in[20];

  // Workspace carve-up (~246 MB, stream-ordered reuse of bufA).
  char* ws = (char*)d_ws;
  size_t off = 0;
  auto carve = [&](size_t bytes) -> void* {
    void* p = ws + off;
    off = (off + bytes + 255) & ~(size_t)255;
    return p;
  };
  unsigned short* bufA = (unsigned short*)carve((size_t)B * KV * HW * 2);
  unsigned short* Kz   = (unsigned short*)carve((size_t)B * KV * HW * 2);
  unsigned short* Vz   = (unsigned short*)carve((size_t)B * KV * HW * 2);
  unsigned short* Qz   = (unsigned short*)carve((size_t)B * 512 * HW * 2);
  float*          Smat = (float*)carve((size_t)B * 512 * KV * 4);
  unsigned short* Pmat = (unsigned short*)carve((size_t)B * 512 * KV * 2);
  float*          rk   = (float*)carve((size_t)B * KV * 4);
  float*          rv   = (float*)carve((size_t)B * KV * 4);   // computed, unused
  float*          rq   = (float*)carve((size_t)B * 512 * 4);
  float*          stats = (float*)carve((size_t)B * 2 * 4);

  const dim3 blk(256);
  const float scale = 1.0f / sqrtf((float)KV);

  // ---- K path: wmk (960x960 GEMM), depthwise 3x3, row norms ----
  wmma_gemm<false, false, false, true><<<dim3(HW / 64, KV / 64, B), blk, 0, stream>>>(
      wmk, emb_all, bufA, KV, KV, HW, HW,
      0LL, (long long)KV * HW, (long long)KV * HW, 1.0f, nullptr, 0, nullptr, 0);
  dw3x3_norm<<<dim3(B * KV), blk, 0, stream>>>(bufA, wk, Kz, rk, KV, 1);

  // ---- V path ----
  wmma_gemm<false, false, false, true><<<dim3(HW / 64, KV / 64, B), blk, 0, stream>>>(
      wmv, emb_all, bufA, KV, KV, HW, HW,
      0LL, (long long)KV * HW, (long long)KV * HW, 1.0f, nullptr, 0, nullptr, 0);
  dw3x3_norm<<<dim3(B * KV), blk, 0, stream>>>(bufA, wv, Vz, rv, KV, 1);

  // ---- Per-branch attention ----
  size_t outOff = 0;
  for (int i = 0; i < 4; ++i) {
    const int c = CNv[i];
    wmma_gemm<false, false, false, true><<<dim3(HW / 64, c / 64, B), blk, 0, stream>>>(
        wm[i], emb[i], bufA, c, c, HW, HW,
        0LL, (long long)c * HW, (long long)c * HW, 1.0f, nullptr, 0, nullptr, 0);
    dw3x3_norm<<<dim3(B * c), blk, 0, stream>>>(bufA, wq[i], Qz, rq, c, 2);
    // S = scale * rq[m] * rk[n] * (Q x K^T)   (N-major B, K=4096)
    wmma_gemm<true, true, true, false><<<dim3(KV / 64, c / 64, B), blk, 0, stream>>>(
        Qz, Kz, Smat, HW, HW, HW, KV,
        (long long)c * HW, (long long)KV * HW, (long long)c * KV,
        scale, rq, c, rk, KV);
    inorm_stats<<<dim3(B), blk, 0, stream>>>(Smat, (long long)c * KV, stats);
    inorm_softmax<<<dim3(B * c), blk, 0, stream>>>(Smat, stats, Pmat, c);
    // O = P x V  (K=960) -> bufA (bf16)
    wmma_gemm<true, true, false, true><<<dim3(HW / 64, c / 64, B), blk, 0, stream>>>(
        Pmat, Vz, bufA, KV, KV, HW, HW,
        (long long)c * KV, (long long)KV * HW, (long long)c * HW,
        1.0f, nullptr, 0, nullptr, 0);
    // out = wp x O  (c x c GEMM), f32 -> d_out
    wmma_gemm<false, true, false, false><<<dim3(HW / 64, c / 64, B), blk, 0, stream>>>(
        wp[i], bufA, (float*)d_out + outOff, c, c, HW, HW,
        0LL, (long long)c * HW, (long long)c * HW, 1.0f, nullptr, 0, nullptr, 0);
    outOff += (size_t)B * c * HW;
  }
}
